// SelfAttention_9844065042689
// MI455X (gfx1250) — matrix operand order repaired and checked by reading, not compiled, not run on hardware
//
#include <hip/hip_runtime.h>
#include <hip/hip_bf16.h>

typedef _Float16 h16 __attribute__((ext_vector_type(16)));
typedef _Float16 h8  __attribute__((ext_vector_type(8)));
typedef float    f8  __attribute__((ext_vector_type(8)));

constexpr int NB    = 4;      // batch
constexpr int NCH   = 256;    // channels C
constexpr int NPOS  = 4096;   // H*W
constexpr int CSPL  = 342;    // ceil(4C/3): used Q/K channels
constexpr int CPAD  = 352;    // 342 padded to multiple of 32
constexpr int MQK   = 704;    // 2*CPAD rows in fused QK weight
constexpr float ATT_SCALE = 0.0625f; // 1/sqrt(256)

union Hfrag { h16 v; h8 h[2]; };

#define WMMA_F16(a, b, c) \
  __builtin_amdgcn_wmma_f32_16x16x32_f16(false, (a), false, (b), (short)0, (c), false, false)

#if __has_builtin(__builtin_amdgcn_tensor_load_to_lds) && __has_builtin(__builtin_amdgcn_s_wait_tensorcnt)
#define USE_TDM 1
#else
#define USE_TDM 0
#endif

#if USE_TDM
typedef unsigned int u32x4 __attribute__((ext_vector_type(4)));
typedef int          i32x4 __attribute__((ext_vector_type(4)));
typedef int          i32x8 __attribute__((ext_vector_type(8)));

// Issue one 2D TDM tile load: tile_d1 rows of tile_d0 f16 elements, global row
// stride stride_d0 elements, packed row-major into LDS at l. (D# per ISA 8.3/8.4)
__device__ __forceinline__ void tdm_issue_2d(const _Float16* g, _Float16* l,
                                             uint32_t tile_d0, uint32_t tile_d1,
                                             uint32_t stride_d0) {
  const uint64_t ga = (uint64_t)(uintptr_t)g;
  const uint32_t la = (uint32_t)(uintptr_t)l;   // low 32 bits of generic == LDS byte addr
  u32x4 g0;
  g0[0] = 1u;                                               // count=1, user descriptor
  g0[1] = la;                                               // lds_addr
  g0[2] = (uint32_t)ga;                                     // global_addr[31:0]
  g0[3] = (uint32_t)((ga >> 32) & 0x01FFFFFFu) | (2u << 30);// global_addr[56:32] | type=2
  i32x8 g1;
  g1[0] = (int)(1u << 16);                                  // wg_mask=0, data_size=1 (2B)
  g1[1] = (int)((tile_d0 & 0xFFFFu) << 16);                 // tensor_dim0[15:0] @ bit 48
  g1[2] = (int)((tile_d0 >> 16) | ((tile_d1 & 0xFFFFu) << 16)); // dim0 hi | tensor_dim1 lo
  g1[3] = (int)((tile_d1 >> 16) | (tile_d0 << 16));         // tensor_dim1 hi | tile_dim0
  g1[4] = (int)(tile_d1 & 0xFFFFu);                         // tile_dim1 (tile_dim2 = 0)
  g1[5] = (int)stride_d0;                                   // tensor_dim0_stride[31:0]
  g1[6] = 0;                                                // stride hi | dim1_stride lo
  g1[7] = 0;
  i32x4 z4 = {};
#if __clang_major__ >= 23
  i32x8 z8 = {};
  __builtin_amdgcn_tensor_load_to_lds(g0, g1, z4, z4, z8, 0);
#else
  __builtin_amdgcn_tensor_load_to_lds(g0, g1, z4, z4, 0);
#endif
}
#endif

// Stage one 32-key chunk (K tile 32xCPAD, V tile NCHx32) into LDS buffers.
__device__ __forceinline__ void stage_chunk(const _Float16* __restrict__ Kb,
                                            const _Float16* __restrict__ Vb,
                                            _Float16* kdst, _Float16* vdst,
                                            int kc, int tid) {
#if USE_TDM
  if (tid < 32) {  // one wave issues the async DMA (TDM ignores EXEC; wave-uniform)
    tdm_issue_2d(Kb + (size_t)kc * CPAD, kdst, CPAD, 32, CPAD);
    tdm_issue_2d(Vb + kc,                vdst, 32, NCH, NPOS);
  }
#else
  const h8* ks = (const h8*)(Kb + (size_t)kc * CPAD);   // K chunk rows are contiguous
  h8* kd = (h8*)kdst;
  for (int i = tid; i < 32 * CPAD / 8; i += 128) kd[i] = ks[i];
  for (int i = tid; i < NCH * 4; i += 128) {
    const int r = i >> 2, c = (i & 3) * 8;
    *(h8*)(vdst + r * 32 + c) = *(const h8*)(Vb + (size_t)r * NPOS + kc + c);
  }
#endif
}

__device__ __forceinline__ void wait_stage(bool more_in_flight) {
#if USE_TDM
  if (more_in_flight) __builtin_amdgcn_s_wait_tensorcnt(2);  // retire current chunk, keep next streaming
  else                __builtin_amdgcn_s_wait_tensorcnt(0);
#endif
  __syncthreads();
}

// ---------------- GroupNorm: block per (b, g); writes norm^T f16 [b][pos][ch] ----
__global__ void gn_kernel(const float* __restrict__ in,
                          const float* __restrict__ gamma,
                          const float* __restrict__ beta,
                          _Float16* __restrict__ normt) {
  const int bg = blockIdx.x, b = bg >> 5, g = bg & 31;
  const float* x = in + ((size_t)b * NCH + g * 8) * NPOS;
  __shared__ float red[256];
  const int tid = threadIdx.x;
  float s = 0.f, ss = 0.f;
  for (int i = tid; i < 8 * NPOS; i += 256) { float v = x[i]; s += v; ss += v * v; }
  red[tid] = s; __syncthreads();
  for (int o = 128; o > 0; o >>= 1) { if (tid < o) red[tid] += red[tid + o]; __syncthreads(); }
  const float mean = red[0] * (1.f / (8 * NPOS));
  __syncthreads();
  red[tid] = ss; __syncthreads();
  for (int o = 128; o > 0; o >>= 1) { if (tid < o) red[tid] += red[tid + o]; __syncthreads(); }
  const float var = red[0] * (1.f / (8 * NPOS)) - mean * mean;
  const float rinv = rsqrtf(var + 1e-5f);
  for (int i = tid; i < 8 * NPOS; i += 256) {
    const int cg = i >> 12, pos = i & (NPOS - 1), ch = g * 8 + cg;
    const float v = (x[i] - mean) * rinv * gamma[ch] + beta[ch];
    normt[((size_t)b * NPOS + pos) * NCH + ch] = (_Float16)v;
  }
}

// ---------------- input -> f16 V [b][ch][pos] (natural layout == B^T for P*V) ---
__global__ void vconv_kernel(const float* __restrict__ in, _Float16* __restrict__ vf) {
  const size_t i = (size_t)blockIdx.x * 256 + threadIdx.x;
  vf[i] = (_Float16)in[i];
}

// ---------------- weight prep: fused QK weight (padded) + bias, out_w -> f16 ----
__global__ void wprep_kernel(const float* __restrict__ qkv_w,
                             const float* __restrict__ qkv_b,
                             const float* __restrict__ out_w,
                             _Float16* __restrict__ wqk,
                             float* __restrict__ bqk,
                             _Float16* __restrict__ wout) {
  const int i = blockIdx.x * 256 + threadIdx.x;
  if (i < MQK * NCH) {
    const int r = i / NCH, c = i - r * NCH;
    int src = -1;
    if (r < CSPL) src = r;                                         // Q rows
    else if (r >= CPAD && r < CPAD + CSPL) src = r - CPAD + CSPL;  // K rows
    wqk[i] = (_Float16)((src >= 0) ? qkv_w[src * NCH + c] : 0.f);
    if (c == 0) bqk[r] = (src >= 0) ? qkv_b[src] : 0.f;
  }
  if (i < NCH * NCH) wout[i] = (_Float16)out_w[i];
}

// ---------------- QKV GEMM: wave per 16x16 tile; D rows -> Q^T/K^T [pos][CPAD] --
__global__ void __launch_bounds__(32)
qkv_gemm_kernel(const _Float16* __restrict__ wqk, const float* __restrict__ bqk,
                const _Float16* __restrict__ normt,
                _Float16* __restrict__ Qt, _Float16* __restrict__ Kt) {
  const int mt = blockIdx.x, nt = blockIdx.y, b = blockIdx.z;
  const int lane = threadIdx.x & 31, col = lane & 15, hi = lane >> 4;
  const _Float16* nb = normt + (size_t)b * NPOS * NCH;
  f8 acc = {};
#pragma unroll
  for (int t = 0; t < 8; ++t) {
    Hfrag a, bb;
    const _Float16* pa = wqk + (size_t)(mt * 16 + col) * NCH + t * 32 + hi * 8;
    a.h[0] = *(const h8*)pa; a.h[1] = *(const h8*)(pa + 16);
    const _Float16* pb = nb + (size_t)(nt * 16 + col) * NCH + t * 32 + hi * 8;
    bb.h[0] = *(const h8*)pb; bb.h[1] = *(const h8*)(pb + 16);
    acc = WMMA_F16(a.v, bb.v, acc);
  }
  const int n = nt * 16 + col;
#pragma unroll
  for (int v = 0; v < 8; ++v) {
    const int m = mt * 16 + v + hi * 8;
    const float val = acc[v] + bqk[m];
    _Float16* dst = (m < CPAD) ? (Qt + ((size_t)b * NPOS + n) * CPAD + m)
                               : (Kt + ((size_t)b * NPOS + n) * CPAD + (m - CPAD));
    *dst = (_Float16)val;
  }
}

// ---------------- flash attention: 4 waves/block, TDM-staged K/V in LDS ---------
__global__ void __launch_bounds__(128)
flash_kernel(const _Float16* __restrict__ Qt, const _Float16* __restrict__ Kt,
             const _Float16* __restrict__ Vf, _Float16* __restrict__ AOt) {
  const int b = blockIdx.y;
  const int tid = threadIdx.x;
  const int wid = tid >> 5, lane = tid & 31, col = lane & 15, hi = lane >> 4;
  const int q0 = blockIdx.x * 64 + wid * 16;
  const _Float16* Qb = Qt + (size_t)b * NPOS * CPAD;
  const _Float16* Kb = Kt + (size_t)b * NPOS * CPAD;
  const _Float16* Vb = Vf + (size_t)b * NCH * NPOS;

  __shared__ _Float16 Klds[2][32 * CPAD];   // 2 x 22 KB
  __shared__ _Float16 Vlds[2][NCH * 32];    // 2 x 16 KB
  __shared__ _Float16 Plds[4][16 * 32];     // per-wave P transpose scratch

  Hfrag qf[11];
#pragma unroll
  for (int t = 0; t < 11; ++t) {
    const _Float16* p = Qb + (size_t)(q0 + col) * CPAD + t * 32 + hi * 16;
    qf[t].h[0] = *(const h8*)p; qf[t].h[1] = *(const h8*)(p + 8);
  }
  f8 O[16];
#pragma unroll
  for (int j = 0; j < 16; ++j) O[j] = (f8){};
  float mrow[8], lrow[8];
#pragma unroll
  for (int v = 0; v < 8; ++v) { mrow[v] = -3.0e38f; lrow[v] = 0.f; }

  stage_chunk(Kb, Vb, Klds[0], Vlds[0], 0, tid);
  stage_chunk(Kb, Vb, Klds[1], Vlds[1], 32, tid);

  for (int kc = 0; kc < NPOS; kc += 32) {
    const int p = (kc >> 5) & 1;
    wait_stage(kc + 64 < NPOS);
    const _Float16* Kp = Klds[p];
    const _Float16* Vp = Vlds[p];

    f8 s0 = {}, s1 = {};
#pragma unroll
    for (int t = 0; t < 11; ++t) {
      Hfrag k0f, k1f;
      const _Float16* p0 = Kp + (size_t)col * CPAD + t * 32 + hi * 16;
      k0f.h[0] = *(const h8*)p0; k0f.h[1] = *(const h8*)(p0 + 8);
      const _Float16* p1 = Kp + (size_t)(16 + col) * CPAD + t * 32 + hi * 16;
      k1f.h[0] = *(const h8*)p1; k1f.h[1] = *(const h8*)(p1 + 8);
      s0 = WMMA_F16(qf[t].v, k0f.v, s0);
      s1 = WMMA_F16(qf[t].v, k1f.v, s1);
    }
    // online softmax per row (row m = v + hi*8; its 16 cols live in the same half-wave)
#pragma unroll
    for (int v = 0; v < 8; ++v) {
      const float a0 = s0[v] * ATT_SCALE, a1 = s1[v] * ATT_SCALE;
      float mx = fmaxf(a0, a1);
#pragma unroll
      for (int m = 1; m < 16; m <<= 1) mx = fmaxf(mx, __shfl_xor(mx, m, 32));
      const float mnew = fmaxf(mrow[v], mx);
      const float corr = __expf(mrow[v] - mnew);
      const float e0 = __expf(a0 - mnew), e1 = __expf(a1 - mnew);
      float rs = e0 + e1;
#pragma unroll
      for (int m = 1; m < 16; m <<= 1) rs += __shfl_xor(rs, m, 32);
      lrow[v] = lrow[v] * corr + rs;
      mrow[v] = mnew;
#pragma unroll
      for (int j = 0; j < 16; ++j) O[j][v] *= corr;
      Plds[wid][(v + hi * 8) * 32 + col]      = (_Float16)e0;
      Plds[wid][(v + hi * 8) * 32 + 16 + col] = (_Float16)e1;
    }
    __syncthreads();
    Hfrag pf;  // re-read P as a 16x32 A-fragment
    const _Float16* pp = &Plds[wid][col * 32 + hi * 16];
    pf.h[0] = *(const h8*)pp; pf.h[1] = *(const h8*)(pp + 8);
#pragma unroll
    for (int j = 0; j < 16; ++j) {
      Hfrag vfb;
      const _Float16* pv = Vp + (size_t)(j * 16 + col) * 32 + hi * 16;
      vfb.h[0] = *(const h8*)pv; vfb.h[1] = *(const h8*)(pv + 8);
      O[j] = WMMA_F16(pf.v, vfb.v, O[j]);
    }
    __syncthreads();  // everyone done with buffer p before it is re-staged
    if (kc + 64 < NPOS) stage_chunk(Kb, Vb, Klds[p], Vlds[p], kc + 64, tid);
  }
#pragma unroll
  for (int v = 0; v < 8; ++v) {
    const float inv = 1.f / lrow[v];
    const int pos = q0 + v + hi * 8;
#pragma unroll
    for (int j = 0; j < 16; ++j)
      AOt[((size_t)b * NPOS + pos) * NCH + j * 16 + col] = (_Float16)(O[j][v] * inv);
  }
}

// ---------------- output projection + bias + residual ---------------------------
__global__ void __launch_bounds__(32)
outproj_kernel(const _Float16* __restrict__ wout, const float* __restrict__ out_b,
               const _Float16* __restrict__ AOt, const float* __restrict__ input,
               float* __restrict__ out) {
  const int mt = blockIdx.x, nt = blockIdx.y, b = blockIdx.z;
  const int lane = threadIdx.x & 31, col = lane & 15, hi = lane >> 4;
  const _Float16* ab = AOt + (size_t)b * NPOS * NCH;
  f8 acc = {};
#pragma unroll
  for (int t = 0; t < 8; ++t) {
    Hfrag a, bb;
    const _Float16* pa = wout + (size_t)(mt * 16 + col) * NCH + t * 32 + hi * 8;
    a.h[0] = *(const h8*)pa; a.h[1] = *(const h8*)(pa + 16);
    const _Float16* pb = ab + (size_t)(nt * 16 + col) * NCH + t * 32 + hi * 8;
    bb.h[0] = *(const h8*)pb; bb.h[1] = *(const h8*)(pb + 16);
    acc = WMMA_F16(a.v, bb.v, acc);
  }
  const int n = nt * 16 + col;
#pragma unroll
  for (int v = 0; v < 8; ++v) {
    const int o = mt * 16 + v + hi * 8;
    const size_t idx = ((size_t)b * NCH + o) * NPOS + n;
    out[idx] = acc[v] + out_b[o] + input[idx];
  }
}

extern "C" void kernel_launch(void* const* d_in, const int* in_sizes, int n_in,
                              void* d_out, int out_size, void* d_ws, size_t ws_size,
                              hipStream_t stream) {
  (void)in_sizes; (void)n_in; (void)out_size; (void)ws_size;
  const float* input = (const float*)d_in[0];
  const float* gamma = (const float*)d_in[1];
  const float* beta  = (const float*)d_in[2];
  const float* qkv_w = (const float*)d_in[3];
  const float* qkv_b = (const float*)d_in[4];
  const float* out_w = (const float*)d_in[5];
  const float* out_b = (const float*)d_in[6];
  float* out = (float*)d_out;

  char* ws = (char*)d_ws;
  size_t off = 0;
  auto carve = [&](size_t bytes) { char* p = ws + off; off = (off + bytes + 255) & ~(size_t)255; return p; };
  _Float16* normt = (_Float16*)carve((size_t)NB * NPOS * NCH * 2);
  _Float16* Qt    = (_Float16*)carve((size_t)NB * NPOS * CPAD * 2);
  _Float16* Kt    = (_Float16*)carve((size_t)NB * NPOS * CPAD * 2);
  _Float16* Vf    = (_Float16*)carve((size_t)NB * NCH * NPOS * 2);
  _Float16* AOt   = (_Float16*)carve((size_t)NB * NPOS * NCH * 2);
  _Float16* Wqk   = (_Float16*)carve((size_t)MQK * NCH * 2);
  float*    Bqk   = (float*)carve((size_t)MQK * 4);
  _Float16* Wout  = (_Float16*)carve((size_t)NCH * NCH * 2);

  gn_kernel<<<NB * 32, 256, 0, stream>>>(input, gamma, beta, normt);
  vconv_kernel<<<(NB * NCH * NPOS) / 256, 256, 0, stream>>>(input, Vf);
  wprep_kernel<<<(MQK * NCH + 255) / 256, 256, 0, stream>>>(qkv_w, qkv_b, out_w, Wqk, Bqk, Wout);
  qkv_gemm_kernel<<<dim3(MQK / 16, NPOS / 16, NB), 32, 0, stream>>>(Wqk, Bqk, normt, Qt, Kt);
  flash_kernel<<<dim3(NPOS / 64, NB), 128, 0, stream>>>(Qt, Kt, Vf, AOt);
  outproj_kernel<<<dim3(NCH / 16, NPOS / 16, NB), 32, 0, stream>>>(Wout, out_b, AOt, input, out);
}
